// GNNStack_20323785244960
// MI455X (gfx1250) — matrix operand order repaired and checked by reading, not compiled, not run on hardware
//
#include <hip/hip_runtime.h>
#include <hip/hip_bf16.h>

typedef __attribute__((ext_vector_type(16))) _Float16 v16h;
typedef __attribute__((ext_vector_type(8)))  _Float16 v8h;
typedef __attribute__((ext_vector_type(4)))  _Float16 h4;
typedef __attribute__((ext_vector_type(8)))  float    v8f;

#define NODES 50000
#define DH 128

// ---------------- WMMA fragment loaders (CDNA5 16x16x32 f16 layouts) ----------------
// A matrix 16x32 (MxK), row-major in memory with row stride K (halves):
//   lanes 0-15: M=lane, K = k0+0..7 / k0+16..23 ; lanes16-31: M=lane-16, K = k0+8..15 / k0+24..31
static __device__ __forceinline__ v16h load_a_frag(const _Float16* __restrict__ A,
                                                   int K, int row0, int k0, int lane) {
    int r  = row0 + (lane & 15);
    int ks = k0 + ((lane >> 4) << 3);
    const _Float16* p = A + (size_t)r * K + ks;
    v8h lo = *(const v8h*)(p);
    v8h hi = *(const v8h*)(p + 16);
    return __builtin_shufflevector(lo, hi, 0,1,2,3,4,5,6,7,8,9,10,11,12,13,14,15);
}

// B matrix 32x16 (KxN) where B = W^T, W staged in LDS as [rows][LP] (LP = K+8 halves):
//   lanes 0-15: N=lane, K = k0+0..7 / k0+16..23 ; lanes16-31: N=lane-16, K = k0+8..15 / k0+24..31
static __device__ __forceinline__ v16h load_b_lds(const _Float16* Wl, int LP,
                                                  int n0, int k0, int lane) {
    int n  = n0 + (lane & 15);
    int ks = k0 + ((lane >> 4) << 3);
    const _Float16* p = Wl + n * LP + ks;
    v8h lo = *(const v8h*)(p);
    v8h hi = *(const v8h*)(p + 16);
    return __builtin_shufflevector(lo, hi, 0,1,2,3,4,5,6,7,8,9,10,11,12,13,14,15);
}

static __device__ __forceinline__ v8f wmma_f16(v16h a, v16h b, v8f c) {
    return __builtin_amdgcn_wmma_f32_16x16x32_f16(false, a, false, b, (short)0, c, false, false);
}

// Cooperative global -> LDS weight staging with +8-half row padding (bank-conflict free B reads)
template<int K>
static __device__ __forceinline__ void fill_lds(_Float16* Wl, const _Float16* __restrict__ W,
                                                int rows) {
    constexpr int LP = K + 8;
    const int chunks = K / 8;
    for (int i = threadIdx.x; i < rows * chunks; i += blockDim.x) {
        int r = i / chunks, c = (i - r * chunks) * 8;
        *(v8h*)(Wl + r * LP + c) = *(const v8h*)(W + r * K + c);
    }
    __syncthreads();
}

template<int K, int NT>
static __device__ __forceinline__ void mma_rows(const _Float16* __restrict__ A,
                                                const _Float16* Wl, int row, int lane, v8f* acc) {
    constexpr int LP = K + 8;
#pragma unroll
    for (int k0 = 0; k0 < K; k0 += 32) {
        v16h a = load_a_frag(A, K, row, k0, lane);
#pragma unroll
        for (int nt = 0; nt < NT; ++nt) {
            v16h b = load_b_lds(Wl, LP, nt * 16, k0, lane);
            acc[nt] = wmma_f16(a, b, acc[nt]);
        }
    }
}

// ---------------- prep kernels ----------------
__global__ void k_wcat(const float* __restrict__ wl, const float* __restrict__ wr,
                       _Float16* __restrict__ out) {
    int i = blockIdx.x * blockDim.x + threadIdx.x;
    if (i >= 128 * 256) return;
    int o = i >> 8, k = i & 255;
    float v = (k < 128) ? wl[o * 128 + k] : wr[o * 128 + (k - 128)];
    out[i] = (_Float16)v;
}

__global__ void k_cvt16(const float* __restrict__ w, _Float16* __restrict__ out, int n) {
    int i = blockIdx.x * blockDim.x + threadIdx.x;
    if (i < n) out[i] = (_Float16)w[i];
}

__global__ void k_wp2pad(const float* __restrict__ wp2, _Float16* __restrict__ out) {
    int i = blockIdx.x * blockDim.x + threadIdx.x;
    if (i >= 48 * 128) return;
    int r = i >> 7;
    out[i] = (_Float16)((r < 40) ? wp2[r * 128 + (i & 127)] : 0.0f);
}

__global__ void k_bias_add(const float* __restrict__ a, const float* __restrict__ b,
                           float* __restrict__ out, int n) {
    int i = blockIdx.x * blockDim.x + threadIdx.x;
    if (i < n) out[i] = a[i] + b[i];
}

__global__ void k_biaspad48(const float* __restrict__ b, float* __restrict__ out) {
    int i = blockIdx.x * blockDim.x + threadIdx.x;
    if (i < 48) out[i] = (i < 40) ? b[i] : 0.0f;
}

__global__ void k_zero(float* __restrict__ p, int n) {
    int i = blockIdx.x * blockDim.x + threadIdx.x;
    if (i < n) p[i] = 0.0f;
}

// ---------------- scatter-mean (edge-parallel atomics; 1 wave = 1 edge row, coalesced) --------
__global__ void k_scatter_f32(const float* __restrict__ x,
                              const long long* __restrict__ src,
                              const long long* __restrict__ dst,
                              float* __restrict__ agg, float* __restrict__ cnt, int E) {
    long long t = (long long)blockIdx.x * blockDim.x + threadIdx.x;
    if (t >= (long long)E * 32) return;
    int e = (int)(t >> 5);
    int c = (int)(t & 31) << 2;
    int s = (int)src[e], d = (int)dst[e];
    float4 v = *(const float4*)(x + (size_t)s * DH + c);
    float* a = agg + (size_t)d * DH + c;
    atomicAdd(a + 0, v.x); atomicAdd(a + 1, v.y);
    atomicAdd(a + 2, v.z); atomicAdd(a + 3, v.w);
    if (c == 0) atomicAdd(cnt + d, 1.0f);
}

__global__ void k_scatter_f16(const _Float16* __restrict__ h,
                              const long long* __restrict__ src,
                              const long long* __restrict__ dst,
                              float* __restrict__ agg, float* __restrict__ cnt, int E) {
    long long t = (long long)blockIdx.x * blockDim.x + threadIdx.x;
    if (t >= (long long)E * 32) return;
    int e = (int)(t >> 5);
    int c = (int)(t & 31) << 2;
    int s = (int)src[e], d = (int)dst[e];
    h4 v = *(const h4*)(h + (size_t)s * DH + c);
    float* a = agg + (size_t)d * DH + c;
    atomicAdd(a + 0, (float)v.x); atomicAdd(a + 1, (float)v.y);
    atomicAdd(a + 2, (float)v.z); atomicAdd(a + 3, (float)v.w);
    if (c == 0) atomicAdd(cnt + d, 1.0f);
}

// ---------------- assemble [feat | mean-agg] as f16, K=256 ----------------
__global__ void k_assemble_f32(const float* __restrict__ feat, const float* __restrict__ agg,
                               const float* __restrict__ cnt, _Float16* __restrict__ xa) {
    int i = blockIdx.x * blockDim.x + threadIdx.x;
    if (i >= NODES * DH) return;
    int node = i >> 7, c = i & 127;
    float ic = 1.0f / fmaxf(cnt[node], 1.0f);
    xa[(size_t)node * 256 + c]       = (_Float16)feat[i];
    xa[(size_t)node * 256 + 128 + c] = (_Float16)(agg[i] * ic);
}

__global__ void k_assemble_f16(const _Float16* __restrict__ feat, const float* __restrict__ agg,
                               const float* __restrict__ cnt, _Float16* __restrict__ xa) {
    int i = blockIdx.x * blockDim.x + threadIdx.x;
    if (i >= NODES * DH) return;
    int node = i >> 7, c = i & 127;
    float ic = 1.0f / fmaxf(cnt[node], 1.0f);
    xa[(size_t)node * 256 + c]       = feat[i];
    xa[(size_t)node * 256 + 128 + c] = (_Float16)(agg[i] * ic);
}

// ---------------- WMMA GEMM: out = normalize_relu(A @ W^T + bias), N_out = 128 --------------
template<int K>
__global__ __launch_bounds__(128, 1)
void k_gemm_norm_relu(const _Float16* __restrict__ A, const _Float16* __restrict__ W,
                      const float* __restrict__ bias, _Float16* __restrict__ Out, int M) {
    extern __shared__ _Float16 smem[];
    fill_lds<K>(smem, W, 128);
    const int lane  = threadIdx.x & 31;
    const int tiles = M >> 4;
    const int nw    = gridDim.x * (blockDim.x >> 5);
    const v8f zero  = {0, 0, 0, 0, 0, 0, 0, 0};
    const int nn    = lane & 15;

    for (int t = blockIdx.x * (blockDim.x >> 5) + (threadIdx.x >> 5); t < tiles; t += nw) {
        int row = t * 16;
        v8f acc[8];
#pragma unroll
        for (int nt = 0; nt < 8; ++nt) acc[nt] = zero;
        mma_rows<K, 8>(A, smem, row, lane, acc);

        float ss[8];
#pragma unroll
        for (int v = 0; v < 8; ++v) ss[v] = 0.0f;
#pragma unroll
        for (int nt = 0; nt < 8; ++nt) {
            float bn = bias[nt * 16 + nn];
#pragma unroll
            for (int v = 0; v < 8; ++v) {
                float x = acc[nt][v] + bn;
                acc[nt][v] = x;
                ss[v] += x * x;
            }
        }
#pragma unroll
        for (int v = 0; v < 8; ++v) {
            float s = ss[v];
#pragma unroll
            for (int off = 1; off < 16; off <<= 1) s += __shfl_xor(s, off, 32);
            ss[v] = 1.0f / fmaxf(sqrtf(s), 1e-12f);
        }
        int rbase = row + ((lane >> 4) << 3);
#pragma unroll
        for (int nt = 0; nt < 8; ++nt)
#pragma unroll
            for (int v = 0; v < 8; ++v) {
                float x = fmaxf(acc[nt][v] * ss[v], 0.0f);
                Out[(size_t)(rbase + v) * DH + nt * 16 + nn] = (_Float16)x;
            }
    }
}

// ---------------- WMMA GEMM: out = A @ W^T + bias (f16 out, N_out=128) ----------------
template<int K>
__global__ __launch_bounds__(128, 1)
void k_gemm_bias(const _Float16* __restrict__ A, const _Float16* __restrict__ W,
                 const float* __restrict__ bias, _Float16* __restrict__ Out, int M) {
    extern __shared__ _Float16 smem[];
    fill_lds<K>(smem, W, 128);
    const int lane  = threadIdx.x & 31;
    const int tiles = M >> 4;
    const int nw    = gridDim.x * (blockDim.x >> 5);
    const v8f zero  = {0, 0, 0, 0, 0, 0, 0, 0};
    const int nn    = lane & 15;

    for (int t = blockIdx.x * (blockDim.x >> 5) + (threadIdx.x >> 5); t < tiles; t += nw) {
        int row = t * 16;
        v8f acc[8];
#pragma unroll
        for (int nt = 0; nt < 8; ++nt) acc[nt] = zero;
        mma_rows<K, 8>(A, smem, row, lane, acc);

        int rbase = row + ((lane >> 4) << 3);
#pragma unroll
        for (int nt = 0; nt < 8; ++nt) {
            float bn = bias[nt * 16 + nn];
#pragma unroll
            for (int v = 0; v < 8; ++v)
                Out[(size_t)(rbase + v) * DH + nt * 16 + nn] = (_Float16)(acc[nt][v] + bn);
        }
    }
}

// ---------------- WMMA GEMM + log_softmax: N_out = 40 (padded 48) ----------------
template<int K>
__global__ __launch_bounds__(128, 1)
void k_gemm_logsm(const _Float16* __restrict__ A, const _Float16* __restrict__ W,
                  const float* __restrict__ bias, float* __restrict__ Out, int M) {
    extern __shared__ _Float16 smem[];
    fill_lds<K>(smem, W, 48);
    const int lane  = threadIdx.x & 31;
    const int tiles = M >> 4;
    const int nw    = gridDim.x * (blockDim.x >> 5);
    const v8f zero  = {0, 0, 0, 0, 0, 0, 0, 0};
    const int nn    = lane & 15;

    for (int t = blockIdx.x * (blockDim.x >> 5) + (threadIdx.x >> 5); t < tiles; t += nw) {
        int row = t * 16;
        v8f acc[3];
#pragma unroll
        for (int nt = 0; nt < 3; ++nt) acc[nt] = zero;
        mma_rows<K, 3>(A, smem, row, lane, acc);

#pragma unroll
        for (int nt = 0; nt < 3; ++nt) {
            float bn = bias[nt * 16 + nn];  // zero-padded past 40
#pragma unroll
            for (int v = 0; v < 8; ++v) acc[nt][v] += bn;
        }
        float mx[8], se[8];
#pragma unroll
        for (int v = 0; v < 8; ++v) {
            float m = -3.0e38f;
#pragma unroll
            for (int nt = 0; nt < 3; ++nt) {
                bool valid = (nt * 16 + nn) < 40;
                m = fmaxf(m, valid ? acc[nt][v] : -3.0e38f);
            }
#pragma unroll
            for (int off = 1; off < 16; off <<= 1) m = fmaxf(m, __shfl_xor(m, off, 32));
            mx[v] = m;
            float s = 0.0f;
#pragma unroll
            for (int nt = 0; nt < 3; ++nt) {
                bool valid = (nt * 16 + nn) < 40;
                s += valid ? __expf(acc[nt][v] - m) : 0.0f;
            }
#pragma unroll
            for (int off = 1; off < 16; off <<= 1) s += __shfl_xor(s, off, 32);
            se[v] = logf(s);
        }
        int rbase = row + ((lane >> 4) << 3);
#pragma unroll
        for (int nt = 0; nt < 3; ++nt) {
            int n = nt * 16 + nn;
            if (n < 40) {
#pragma unroll
                for (int v = 0; v < 8; ++v)
                    Out[(size_t)(rbase + v) * 40 + n] = acc[nt][v] - mx[v] - se[v];
            }
        }
    }
}

// ---------------- host side ----------------
extern "C" void kernel_launch(void* const* d_in, const int* in_sizes, int n_in,
                              void* d_out, int out_size, void* d_ws, size_t ws_size,
                              hipStream_t stream) {
    const float*     x    = (const float*)d_in[0];
    const long long* ei   = (const long long*)d_in[1];
    const float*     w_l0 = (const float*)d_in[2];
    const float*     b_l0 = (const float*)d_in[3];
    const float*     w_r0 = (const float*)d_in[4];
    const float*     b_r0 = (const float*)d_in[5];
    const float*     w_l1 = (const float*)d_in[6];
    const float*     b_l1 = (const float*)d_in[7];
    const float*     w_r1 = (const float*)d_in[8];
    const float*     b_r1 = (const float*)d_in[9];
    const float*     wp1  = (const float*)d_in[10];
    const float*     bp1  = (const float*)d_in[11];
    const float*     wp2  = (const float*)d_in[12];
    const float*     bp2  = (const float*)d_in[13];

    const int E = in_sizes[1] / 2;
    const long long* src = ei;
    const long long* dst = ei + E;

    // workspace carve-out (256B aligned)
    char* base = (char*)d_ws;
    size_t off = 0;
    auto carve = [&](size_t bytes) -> char* {
        char* p = base + off;
        off += (bytes + 255) & ~(size_t)255;
        return p;
    };
    float*    agg   = (float*)carve((size_t)NODES * DH * 4 + NODES * 4);  // agg + cnt contiguous
    float*    cnt   = agg + (size_t)NODES * DH;
    _Float16* xa    = (_Float16*)carve((size_t)NODES * 256 * 2);
    _Float16* h1    = (_Float16*)carve((size_t)NODES * DH * 2);
    _Float16* h2    = (_Float16*)carve((size_t)NODES * DH * 2);
    _Float16* z1    = (_Float16*)carve((size_t)NODES * DH * 2);
    _Float16* wcat0 = (_Float16*)carve(128 * 256 * 2);
    _Float16* wcat1 = (_Float16*)carve(128 * 256 * 2);
    _Float16* wp1h  = (_Float16*)carve(128 * 128 * 2);
    _Float16* wp2p  = (_Float16*)carve(48 * 128 * 2);
    float*    bb0   = (float*)carve(128 * 4);
    float*    bb1   = (float*)carve(128 * 4);
    float*    bp2p  = (float*)carve(48 * 4);

    const int TPB = 256;
    const int nElem  = NODES * DH;                       // 6.4M
    const int nEdgeT = (int)(((long long)E * 32 + TPB - 1) / TPB);
    const int gElem  = (nElem + TPB - 1) / TPB;
    const int gemmB  = 128;                              // 4 waves/block, grid-stride tiles
    const size_t sh256 = (size_t)128 * (256 + 8) * 2;    // 67584 B
    const size_t sh128 = (size_t)128 * (128 + 8) * 2;    // 34816 B
    const size_t sh48  = (size_t)48  * (128 + 8) * 2;    // 13056 B

    // ---- weight / bias prep ----
    k_wcat   <<<(128 * 256 + TPB - 1) / TPB, TPB, 0, stream>>>(w_l0, w_r0, wcat0);
    k_wcat   <<<(128 * 256 + TPB - 1) / TPB, TPB, 0, stream>>>(w_l1, w_r1, wcat1);
    k_cvt16  <<<(128 * 128 + TPB - 1) / TPB, TPB, 0, stream>>>(wp1, wp1h, 128 * 128);
    k_wp2pad <<<(48 * 128 + TPB - 1) / TPB, TPB, 0, stream>>>(wp2, wp2p);
    k_bias_add<<<1, 128, 0, stream>>>(b_l0, b_r0, bb0, 128);
    k_bias_add<<<1, 128, 0, stream>>>(b_l1, b_r1, bb1, 128);
    k_biaspad48<<<1, 64, 0, stream>>>(bp2, bp2p);

    // ---- layer 0 ----
    k_zero<<<(NODES * 129 + TPB - 1) / TPB, TPB, 0, stream>>>(agg, NODES * 129);
    k_scatter_f32<<<nEdgeT, TPB, 0, stream>>>(x, src, dst, agg, cnt, E);
    k_assemble_f32<<<gElem, TPB, 0, stream>>>(x, agg, cnt, xa);
    k_gemm_norm_relu<256><<<gemmB, 128, sh256, stream>>>(xa, wcat0, bb0, h1, NODES);

    // ---- layer 1 ----
    k_zero<<<(NODES * 129 + TPB - 1) / TPB, TPB, 0, stream>>>(agg, NODES * 129);
    k_scatter_f16<<<nEdgeT, TPB, 0, stream>>>(h1, src, dst, agg, cnt, E);
    k_assemble_f16<<<gElem, TPB, 0, stream>>>(h1, agg, cnt, xa);
    k_gemm_norm_relu<256><<<gemmB, 128, sh256, stream>>>(xa, wcat1, bb1, h2, NODES);

    // ---- head ----
    k_gemm_bias<128> <<<gemmB, 128, sh128, stream>>>(h2, wp1h, bp1, z1, NODES);
    k_gemm_logsm<128><<<gemmB, 128, sh48, stream>>>(z1, wp2p, bp2p, (float*)d_out, NODES);
}
